// Attention_6682969112960
// MI455X (gfx1250) — matrix operand order repaired and checked by reading, not compiled, not run on hardware
//
#include <hip/hip_runtime.h>

// ---------------------------------------------------------------------------
// MI455X (gfx1250, wave32) fused attention:
//   qkv = x @ Wqkv + b ; flash-attention per head ; out = attn @ Wproj + b
// Matmuls via v_wmma_f32_16x16x32_bf16 (fp32 accumulate); row-major LDS
// staging via GLOBAL_LOAD_ASYNC_TO_LDS_B128 (ASYNCcnt path).
// ---------------------------------------------------------------------------

#define B_  2
#define S_  2048
#define D_  1024
#define H_  16
#define HD_ 64
#define SCALE_ 0.125f  // 64^-0.5

typedef __attribute__((ext_vector_type(16))) __bf16 v16bf;
typedef __attribute__((ext_vector_type(8)))  __bf16 v8bf;
typedef __attribute__((ext_vector_type(8)))  float  v8f;

static __device__ __forceinline__ v8f wmma_bf16(v16bf a, v16bf b, v8f c) {
    // (neg_a, A, neg_b, B, c_mod, C, reuse_a, reuse_b)
    return __builtin_amdgcn_wmma_f32_16x16x32_bf16(false, a, false, b,
                                                   (short)0, c, false, false);
}

// Async 16-byte memory->LDS copy (ISA §15.18.3 opcode 98, tracked by ASYNCcnt).
// `lds` is the LDS byte offset = low 32 bits of the generic shared-pointer
// (generic LDS addr = {SHARED_BASE[63:32], offset[31:0]}, ISA §10.2).
static __device__ __forceinline__ void async_copy16(unsigned lds, const void* g) {
    asm volatile("global_load_async_to_lds_b128 %0, %1, off"
                 :: "v"(lds), "v"((unsigned long long)g)
                 : "memory");
}
static __device__ __forceinline__ unsigned lds_off(const void* p) {
    return (unsigned)(unsigned long long)p;
}
static __device__ __forceinline__ void wait_async0() {
#if __has_builtin(__builtin_amdgcn_s_wait_asynccnt)
    __builtin_amdgcn_s_wait_asynccnt(0);
#else
    asm volatile("s_wait_asynccnt 0x0" ::: "memory");
#endif
}

static __device__ __forceinline__ v16bf combine16(v8bf lo, v8bf hi) {
    v16bf r;
#pragma unroll
    for (int i = 0; i < 8; ++i) { r[i] = lo[i]; r[i + 8] = hi[i]; }
    return r;
}

// A-matrix 16x32 bf16 fragment (ISA 7.12.2): lane m=lane%16, half=lane/16.
// element i -> K = (i>=8?16:0) + half*8 + i%8  => two contiguous 8-elem runs.
static __device__ __forceinline__ v16bf load_afrag(const __bf16* row, int half) {
    v8bf lo = *(const v8bf*)(row + half * 8);
    v8bf hi = *(const v8bf*)(row + 16 + half * 8);
    return combine16(lo, hi);
}

// B-matrix 32x16 bf16 fragment: lane n=lane%16, element i -> K = (i>=8?16:0) + half*8 + i%8.
static __device__ __forceinline__ v16bf load_bfrag(const __bf16* row, int half) {
    v8bf lo = *(const v8bf*)(row + half * 8);
    v8bf hi = *(const v8bf*)(row + 16 + half * 8);
    return combine16(lo, hi);
}

// ---------------------------------------------------------------------------
// fp32 -> bf16 cast (sizes here are all multiples of 4)
// ---------------------------------------------------------------------------
__global__ void cvt_f32_bf16(const float* __restrict__ in,
                             __bf16* __restrict__ out, int n4) {
    int i = blockIdx.x * blockDim.x + threadIdx.x;
    if (i < n4) {
        float4 v = ((const float4*)in)[i];
        __bf16* o = out + i * 4;
        o[0] = (__bf16)v.x; o[1] = (__bf16)v.y;
        o[2] = (__bf16)v.z; o[3] = (__bf16)v.w;
    }
}

// ---------------------------------------------------------------------------
// bf16 GEMM: C[M,N] = A[M,K] @ W[K,N] + bias.  256 thr = 8 waves (2x4),
// block tile 128x128x32, wave tile 64x32 = 4x2 WMMA tiles.
// A tile staged via async-to-LDS; W tile transposed manually.
// ---------------------------------------------------------------------------
template <bool OUT_BF16>
__global__ __launch_bounds__(256)
void gemm_bf16(const __bf16* __restrict__ A, const __bf16* __restrict__ W,
               const float* __restrict__ bias, void* __restrict__ Cout,
               int M, int N, int K) {
    constexpr int BM = 128, BN = 128, BK = 32;
    constexpr int AS = 40;  // LDS stride (elems) for A tile  (80B rows, 16B aligned)
    constexpr int WS = 48;  // LDS stride (elems) for W tile  (96B rows, 16B aligned)
    __shared__ __bf16 Asm[BM * AS];
    __shared__ __bf16 Wsm[BN * WS];  // stored transposed: Wsm[n][k]

    const int tid  = threadIdx.x;
    const int wave = tid >> 5;
    const int lane = tid & 31;
    const int half = lane >> 4;
    const int l16  = lane & 15;
    const int wrow = (wave >> 2) * 64;  // 0 / 64
    const int wcol = (wave & 3) * 32;   // 0 / 32 / 64 / 96
    const int bm = blockIdx.y * BM;
    const int bn = blockIdx.x * BN;

    v8f acc[4][2] = {};

    for (int k0 = 0; k0 < K; k0 += BK) {
        // --- stage A tile 128x32: async memory->LDS, 16B per lane per iter ---
#pragma unroll
        for (int j = 0; j < 2; ++j) {
            int e = j * 2048 + tid * 8;
            int r = e >> 5, c = e & 31;
            async_copy16(lds_off(&Asm[r * AS + c]),
                         A + (size_t)(bm + r) * K + k0 + c);
        }
        // --- stage W tile 32x128, transposed into Wsm[n][k] ---
#pragma unroll
        for (int j = 0; j < 2; ++j) {
            int e = j * 2048 + tid * 8;
            int kk = e >> 7, nn = e & 127;
            uint4 v = *(const uint4*)(W + (size_t)(k0 + kk) * N + bn + nn);
            const __bf16* ve = (const __bf16*)&v;
#pragma unroll
            for (int q = 0; q < 8; ++q) Wsm[(nn + q) * WS + kk] = ve[q];
        }
        if (k0 + BK < K) {  // gfx1250 global_prefetch_b8 for next tiles
            __builtin_prefetch(A + (size_t)(bm + (tid >> 1)) * K + k0 + BK, 0, 1);
            __builtin_prefetch(W + (size_t)(k0 + BK + (tid >> 3)) * N + bn + (tid & 7) * 16, 0, 1);
        }
        wait_async0();
        __syncthreads();

        v16bf aF[4], bF[2];
#pragma unroll
        for (int i = 0; i < 4; ++i)
            aF[i] = load_afrag(&Asm[(wrow + i * 16 + l16) * AS], half);
#pragma unroll
        for (int j = 0; j < 2; ++j)
            bF[j] = load_bfrag(&Wsm[(wcol + j * 16 + l16) * WS], half);
#pragma unroll
        for (int i = 0; i < 4; ++i)
#pragma unroll
            for (int j = 0; j < 2; ++j)
                acc[i][j] = wmma_bf16(aF[i], bF[j], acc[i][j]);
        __syncthreads();
    }

    // --- epilogue: C/D layout row M = r + half*8, col N = l16 ---
#pragma unroll
    for (int i = 0; i < 4; ++i) {
        int row0 = bm + wrow + i * 16 + half * 8;
#pragma unroll
        for (int j = 0; j < 2; ++j) {
            int col = bn + wcol + j * 16 + l16;
            float bv = bias[col];
#pragma unroll
            for (int r = 0; r < 8; ++r) {
                float v = acc[i][j][r] + bv;
                size_t idx = (size_t)(row0 + r) * N + col;
                if constexpr (OUT_BF16) ((__bf16*)Cout)[idx] = (__bf16)v;
                else                    ((float*)Cout)[idx]  = v;
            }
        }
    }
}

// ---------------------------------------------------------------------------
// Flash attention. Block = 128 thr = 4 waves; each wave owns 16 q rows of a
// 64-row q tile for one (b,h). Streams 64-key K/V tiles through LDS with
// online softmax; O accumulated in fp32 registers.
// qkv layout: [B*S, 3*D], cols = c3*D + h*HD + d.
// ---------------------------------------------------------------------------
__global__ __launch_bounds__(128)
void flash_attn(const __bf16* __restrict__ qkv, __bf16* __restrict__ attn_out) {
    constexpr int QS = 72;  // LDS row stride (144B, 16B aligned, conflict-free-ish)
    __shared__ __bf16 Qs[64 * QS];
    __shared__ __bf16 Ks[64 * QS];   // Ks[key][d]
    __shared__ __bf16 Vt[64 * QS];   // Vt[d][key]  (transposed for B-frags)
    __shared__ __bf16 Ps[64 * QS];   // Ps[q][key]

    const int tid  = threadIdx.x;
    const int wave = tid >> 5;
    const int lane = tid & 31;
    const int half = lane >> 4;
    const int l16  = lane & 15;
    const int qt = blockIdx.x, h = blockIdx.y, b = blockIdx.z;

    const size_t rs = 3 * D_;  // qkv row stride
    const __bf16* qbase = qkv + (size_t)b * S_ * rs + h * HD_;
    const __bf16* kbase = qbase + D_;
    const __bf16* vbase = qbase + 2 * D_;

    // --- load Q tile (64 rows x 64 d) into LDS: async copies, 16B chunks ---
    for (int idx = tid; idx < 64 * 8; idx += 128) {
        int r = idx >> 3, c8 = (idx & 7) * 8;
        async_copy16(lds_off(&Qs[r * QS + c8]),
                     qbase + (size_t)(qt * 64 + r) * rs + c8);
    }
    wait_async0();
    __syncthreads();

    // Q A-fragments, kept in registers for the whole key loop
    const __bf16* qrow = &Qs[(wave * 16 + l16) * QS];
    v16bf qA0 = load_afrag(qrow, half);       // d 0..31
    v16bf qA1 = load_afrag(qrow + 32, half);  // d 32..63

    v8f o[4] = {};
    float mrow[8], lrow[8];
#pragma unroll
    for (int r = 0; r < 8; ++r) { mrow[r] = -1e30f; lrow[r] = 0.f; }

    for (int kt = 0; kt < S_ / 64; ++kt) {
        // --- stage K (async, row-major) and V (transposed via VGPR) tiles ---
        for (int idx = tid; idx < 64 * 8; idx += 128) {
            int r = idx >> 3, c8 = (idx & 7) * 8;
            size_t grow = (size_t)(kt * 64 + r) * rs;
            async_copy16(lds_off(&Ks[r * QS + c8]), kbase + grow + c8);
            uint4 vv = *(const uint4*)(vbase + grow + c8);
            const __bf16* ve = (const __bf16*)&vv;
#pragma unroll
            for (int q = 0; q < 8; ++q) Vt[(c8 + q) * QS + r] = ve[q];
        }
        wait_async0();
        __syncthreads();

        // --- S = scale * Q K^T : 4 tiles of 16x16 per wave ---
        v8f st[4];
#pragma unroll
        for (int ct = 0; ct < 4; ++ct) {
            const __bf16* krow = &Ks[(ct * 16 + l16) * QS];
            v8f c = {};
            c = wmma_bf16(qA0, load_bfrag(krow, half), c);
            c = wmma_bf16(qA1, load_bfrag(krow + 32, half), c);
            st[ct] = c * SCALE_;
        }

        // --- online softmax (stats per row, reduced over 16-lane groups) ---
#pragma unroll
        for (int r = 0; r < 8; ++r) {
            float mx = st[0][r];
#pragma unroll
            for (int ct = 1; ct < 4; ++ct) mx = fmaxf(mx, st[ct][r]);
#pragma unroll
            for (int m = 8; m; m >>= 1) mx = fmaxf(mx, __shfl_xor(mx, m, 32));
            float nm = fmaxf(mrow[r], mx);
            float alpha = __expf(mrow[r] - nm);
            mrow[r] = nm;
            float sum = 0.f;
#pragma unroll
            for (int ct = 0; ct < 4; ++ct) {
                float p = __expf(st[ct][r] - nm);
                st[ct][r] = p;
                sum += p;
            }
#pragma unroll
            for (int m = 8; m; m >>= 1) sum += __shfl_xor(sum, m, 32);
            lrow[r] = lrow[r] * alpha + sum;
#pragma unroll
            for (int ct = 0; ct < 4; ++ct) o[ct][r] *= alpha;
        }

        // --- stage P (bf16) for the P·V WMMA ---
#pragma unroll
        for (int ct = 0; ct < 4; ++ct)
#pragma unroll
            for (int r = 0; r < 8; ++r)
                Ps[(wave * 16 + r + half * 8) * QS + ct * 16 + l16] =
                    (__bf16)st[ct][r];
        __syncthreads();

        // --- O += P V ---
        const __bf16* prow = &Ps[(wave * 16 + l16) * QS];
        v16bf pA0 = load_afrag(prow, half);       // keys 0..31
        v16bf pA1 = load_afrag(prow + 32, half);  // keys 32..63
#pragma unroll
        for (int ct = 0; ct < 4; ++ct) {
            const __bf16* vrow = &Vt[(ct * 16 + l16) * QS];  // fixed d column
            o[ct] = wmma_bf16(pA0, load_bfrag(vrow, half), o[ct]);
            o[ct] = wmma_bf16(pA1, load_bfrag(vrow + 32, half), o[ct]);
        }
        __syncthreads();  // before next tile overwrites Ks/Vt
    }

    // --- finalize: O / l, write bf16 [B*S, D] for the projection GEMM ---
#pragma unroll
    for (int ct = 0; ct < 4; ++ct)
#pragma unroll
        for (int r = 0; r < 8; ++r) {
            int qr  = qt * 64 + wave * 16 + r + half * 8;
            int col = h * HD_ + ct * 16 + l16;
            attn_out[((size_t)b * S_ + qr) * D_ + col] =
                (__bf16)(o[ct][r] / lrow[r]);
        }
}

// ---------------------------------------------------------------------------
extern "C" void kernel_launch(void* const* d_in, const int* in_sizes, int n_in,
                              void* d_out, int out_size, void* d_ws, size_t ws_size,
                              hipStream_t stream) {
    (void)in_sizes; (void)n_in; (void)out_size; (void)ws_size;
    const float* x     = (const float*)d_in[0];
    const float* Wqkv  = (const float*)d_in[1];
    const float* bqkv  = (const float*)d_in[2];
    const float* Wproj = (const float*)d_in[3];
    const float* bproj = (const float*)d_in[4];
    float* out = (float*)d_out;

    const size_t nX  = (size_t)B_ * S_ * D_;       // 4,194,304
    const size_t nWq = (size_t)D_ * 3 * D_;        // 3,145,728
    const size_t nWp = (size_t)D_ * D_;            // 1,048,576
    const size_t nQKV = (size_t)B_ * S_ * 3 * D_;  // 12,582,912

    __bf16* xb    = (__bf16*)d_ws;
    __bf16* wqkvb = xb + nX;
    __bf16* wprojb = wqkvb + nWq;
    __bf16* qkvb  = wprojb + nWp;
    __bf16* attnb = qkvb + nQKV;   // total ws use ~50.3 MB

    // fp32 -> bf16 casts
    cvt_f32_bf16<<<(int)(nX / 4 + 255) / 256, 256, 0, stream>>>(x, xb, (int)(nX / 4));
    cvt_f32_bf16<<<(int)(nWq / 4 + 255) / 256, 256, 0, stream>>>(Wqkv, wqkvb, (int)(nWq / 4));
    cvt_f32_bf16<<<(int)(nWp / 4 + 255) / 256, 256, 0, stream>>>(Wproj, wprojb, (int)(nWp / 4));

    // GEMM1: [4096,1024] @ [1024,3072] -> bf16 qkv
    dim3 g1(3 * D_ / 128, B_ * S_ / 128);
    gemm_bf16<true><<<g1, 256, 0, stream>>>(xb, wqkvb, bqkv, qkvb,
                                            B_ * S_, 3 * D_, D_);

    // flash attention
    dim3 ga(S_ / 64, H_, B_);
    flash_attn<<<ga, 128, 0, stream>>>(qkvb, attnb);

    // GEMM2: [4096,1024] @ [1024,1024] + bias -> fp32 out
    dim3 g2(D_ / 128, B_ * S_ / 128);
    gemm_bf16<false><<<g2, 256, 0, stream>>>(attnb, wprojb, bproj, out,
                                             B_ * S_, D_, D_);
}
